// SelectiveSSM_65420941853160
// MI455X (gfx1250) — matrix operand order repaired and checked by reading, not compiled, not run on hardware
//
#include <hip/hip_runtime.h>
#include <hip/hip_bf16.h>
#include <math.h>

// ---------------------------------------------------------------------------
// Problem constants (match reference)
// ---------------------------------------------------------------------------
#define BB 2
#define LL 1024
#define D_MODEL 1024
#define D_STATE 16
#define D_CONV 4
#define D_INNER 2048           // D_MODEL * EXPAND
#define MROWS (BB * LL)        // 2048
#define N_IN (2 * D_INNER)     // 4096
#define NPROJ (2 * D_STATE + 1) // 33
#define LN_EPS 1e-5f

typedef __attribute__((ext_vector_type(16))) __bf16 v16bf;
typedef __attribute__((ext_vector_type(8)))  float  v8f;
typedef int v4i __attribute__((vector_size(16)));   // matches builtin param type

union AFrag { v16bf v; uint4 u[2]; };

#define ASG __attribute__((address_space(1)))
#define ASL __attribute__((address_space(3)))

#if __has_builtin(__builtin_amdgcn_global_load_async_to_lds_b128)
#define HAVE_ASYNC_LDS 1
#else
#define HAVE_ASYNC_LDS 0
#endif

template <int N>
__device__ __forceinline__ void wait_async() {
#if HAVE_ASYNC_LDS
#if __has_builtin(__builtin_amdgcn_s_wait_asynccnt)
    __builtin_amdgcn_s_wait_asynccnt(N);
#else
    asm volatile("s_wait_asynccnt %0" ::"n"(N) : "memory");
#endif
#endif
}

// f32 -> bf16 round-to-nearest-even (bit math; no scalar __bf16 arithmetic)
__device__ __forceinline__ unsigned short f2bf(float f) {
    unsigned int u = __float_as_uint(f);
    unsigned int r = u + 0x7fffu + ((u >> 16) & 1u);
    return (unsigned short)(r >> 16);
}

__device__ __forceinline__ float sigmoidf_(float x) { return 1.0f / (1.0f + expf(-x)); }
__device__ __forceinline__ float siluf_(float x)    { return x * sigmoidf_(x); }
__device__ __forceinline__ float softplusf_(float x) {
    return (x > 20.0f) ? x : log1pf(expf(x));
}

// ---------------------------------------------------------------------------
// f32 -> bf16 conversion
// ---------------------------------------------------------------------------
__global__ __launch_bounds__(256) void cvt_bf16_kernel(const float* __restrict__ in,
                                                       unsigned short* __restrict__ out,
                                                       int n) {
    int i = blockIdx.x * 256 + threadIdx.x;
    if (i < n) out[i] = f2bf(in[i]);
}

// ---------------------------------------------------------------------------
// bf16 WMMA GEMM:  D[m,n] = sum_k A[m,k] * W[n,k]
// A: M x K row-major bf16 (as ushort), W: N x K row-major bf16, D: M x N f32.
// Block = 128 threads = 4 waves stacked in M (block tile 64x64).
// The 64x32 B tile is shared by all 4 waves -> staged into LDS once per
// K-step via async global->LDS (double buffered), cutting B traffic 4x.
// LDS rows padded to 80 B so a fragment's 16 rows hit 16 distinct banks.
// M % 64 == 0, N % 64 == 0, K % 32 == 0.
// ---------------------------------------------------------------------------
#define BROW 40   // ushorts per LDS row (32 data + 8 pad = 80 bytes)

__global__ __launch_bounds__(128) void gemm_bf16_wmma_kernel(
    const unsigned short* __restrict__ A,
    const unsigned short* __restrict__ W,
    float* __restrict__ D,
    int M, int N, int K)
{
    __shared__ __align__(16) unsigned short sB[2][64 * BROW];

    const int wave = threadIdx.x >> 5;
    const int lane = threadIdx.x & 31;
    const int row  = lane & 15;       // M-row within A tile / N-col within B tile
    const int half = lane >> 4;       // K-half selector

    const int m0 = blockIdx.y * 64 + wave * 16;
    const int n0 = blockIdx.x * 64;

    v8f zero = {0.f, 0.f, 0.f, 0.f, 0.f, 0.f, 0.f, 0.f};
    v8f acc0 = zero, acc1 = zero, acc2 = zero, acc3 = zero;

    const unsigned short* arow = A + (size_t)(m0 + row) * K;

    // Stage one 64(N) x 32(K) bf16 tile into sB[buf]:
    // 256 chunks of 16B; each thread moves chunks tid and tid+128.
    auto stage = [&](int buf, int k0) {
        #pragma unroll
        for (int cc = 0; cc < 2; ++cc) {
            int c   = threadIdx.x + cc * 128;
            int r   = c >> 2;        // n_local (0..63)
            int blk = c & 3;         // 16B block within the 64B row
            const unsigned short* gp = W + (size_t)(n0 + r) * K + k0 + blk * 8;
            unsigned short* lp = &sB[buf][r * BROW + blk * 8];
#if HAVE_ASYNC_LDS
            __builtin_amdgcn_global_load_async_to_lds_b128(
                (ASG v4i*)gp, (ASL v4i*)lp, 0, 0);
#else
            *(uint4*)lp = *(const uint4*)gp;
#endif
        }
    };

    const int ksteps = K >> 5;
    stage(0, 0);

    for (int t = 0; t < ksteps; ++t) {
        const int cur = t & 1;
        const int k0  = t << 5;

        if (t + 1 < ksteps) {
            stage(cur ^ 1, k0 + 32);
            wait_async<2>();   // allow next tile in flight; current tile landed
        } else {
            wait_async<0>();
        }
        __syncthreads();       // all waves' chunks of the current tile visible

        // A fragment (16x32 bf16): lane holds K runs [k0+half*8,+8) and [k0+16+half*8,+8)
        AFrag a;
        const unsigned short* ap = arow + k0 + half * 8;
        a.u[0] = *(const uint4*)(ap);
        a.u[1] = *(const uint4*)(ap + 16);

        // B fragments from LDS: lane = column n (row&15), K runs [half*8,+8) and [16+half*8,+8)
        #pragma unroll
        for (int i = 0; i < 4; ++i) {
            AFrag b;
            const unsigned short* bp = &sB[cur][(i * 16 + row) * BROW + half * 8];
            b.u[0] = *(const uint4*)(bp);
            b.u[1] = *(const uint4*)(bp + 16);
            v8f* accp = (i == 0) ? &acc0 : (i == 1) ? &acc1 : (i == 2) ? &acc2 : &acc3;
            *accp = __builtin_amdgcn_wmma_f32_16x16x32_bf16(
                false, a.v, false, b.v, (short)0, *accp, false, false);
        }
        __syncthreads();       // all reads of this buffer done before it is re-staged
    }

    // D 16x16 f32 layout: VGPR r -> M = r + half*8 ; lane&15 -> N
    #pragma unroll
    for (int i = 0; i < 4; ++i) {
        v8f acc = (i == 0) ? acc0 : (i == 1) ? acc1 : (i == 2) ? acc2 : acc3;
        const int n = n0 + i * 16 + row;
        #pragma unroll
        for (int r = 0; r < 8; ++r) {
            D[(size_t)(m0 + r + half * 8) * N + n] = acc[r];
        }
    }
}

// ---------------------------------------------------------------------------
// Causal depthwise conv (D_CONV=4) + SiLU on the x-half of xz.
// xz: [M, 4096] f32 (first 2048 = xb-input, last 2048 = z). xb_out: [M, 2048].
// ---------------------------------------------------------------------------
__global__ __launch_bounds__(256) void conv_silu_kernel(const float* __restrict__ xz,
                                                        const float* __restrict__ conv_w,
                                                        const float* __restrict__ conv_b,
                                                        float* __restrict__ xb_out)
{
    int idx = blockIdx.x * 256 + threadIdx.x;          // over M * D_INNER
    int d  = idx & (D_INNER - 1);
    int ml = idx >> 11;                                 // row index m = b*L + l
    int l  = ml & (LL - 1);

    float acc = conv_b[d];
    #pragma unroll
    for (int k = 0; k < D_CONV; ++k) {
        int li = l + k - (D_CONV - 1);
        if (li >= 0) {
            acc += xz[(size_t)(ml + k - (D_CONV - 1)) * N_IN + d] * conv_w[d * D_CONV + k];
        }
    }
    xb_out[idx] = siluf_(acc);
}

// ---------------------------------------------------------------------------
// x_proj: proj[m,e] = sum_k xb[m,k] * x_proj_w[e,k], e < 33.
// One block per row m; xb row staged in LDS; 8-way split-K per output.
// ---------------------------------------------------------------------------
__global__ __launch_bounds__(256) void xproj_kernel(const float* __restrict__ xb,
                                                    const float* __restrict__ xpw,
                                                    float* __restrict__ proj)
{
    __shared__ float s_x[D_INNER];
    __shared__ float s_part[NPROJ * 8];
    const int m = blockIdx.x;

    for (int k = threadIdx.x; k < D_INNER; k += 256)
        s_x[k] = xb[(size_t)m * D_INNER + k];
    __syncthreads();

    for (int idx = threadIdx.x; idx < NPROJ * 8; idx += 256) {
        int e = idx >> 3, part = idx & 7;
        const float* w = xpw + (size_t)e * D_INNER + part * 256;
        const float* xv = s_x + part * 256;
        float acc = 0.f;
        #pragma unroll 8
        for (int k = 0; k < 256; ++k) acc += xv[k] * w[k];
        s_part[idx] = acc;
    }
    __syncthreads();

    if (threadIdx.x < NPROJ) {
        float acc = 0.f;
        #pragma unroll
        for (int p = 0; p < 8; ++p) acc += s_part[threadIdx.x * 8 + p];
        proj[(size_t)m * NPROJ + threadIdx.x] = acc;
    }
}

// ---------------------------------------------------------------------------
// Selective scan: one thread per (b,d); 16-entry state in registers.
// Per-step shared proj row (dt_in, B[16], C[16]) broadcast via LDS.
// y_raw[b,l,d] = sum_n h[n]*C[n] + xb*D_param
// ---------------------------------------------------------------------------
__global__ __launch_bounds__(256) void scan_kernel(const float* __restrict__ xb,
                                                   const float* __restrict__ proj,
                                                   const float* __restrict__ A_log,
                                                   const float* __restrict__ D_param,
                                                   const float* __restrict__ dt_w,
                                                   const float* __restrict__ dt_b,
                                                   float* __restrict__ y_raw)
{
    const int b = blockIdx.y;
    const int d = blockIdx.x * 256 + threadIdx.x;

    __shared__ float s_proj[NPROJ];

    float A[D_STATE];
    #pragma unroll
    for (int n = 0; n < D_STATE; ++n) A[n] = -expf(A_log[d * D_STATE + n]);
    const float dtw = dt_w[d];
    const float dtb = dt_b[d];
    const float Dp  = D_param[d];

    float h[D_STATE];
    #pragma unroll
    for (int n = 0; n < D_STATE; ++n) h[n] = 0.f;

    const float* pb  = proj + (size_t)b * LL * NPROJ;
    const float* xbb = xb   + (size_t)b * LL * D_INNER;
    float*       yb  = y_raw + (size_t)b * LL * D_INNER;

    for (int l = 0; l < LL; ++l) {
        if (threadIdx.x < NPROJ) s_proj[threadIdx.x] = pb[l * NPROJ + threadIdx.x];
        __syncthreads();

        const float xv = xbb[(size_t)l * D_INNER + d];
        const float dt = softplusf_(s_proj[0] * dtw + dtb);
        const float xd = xv * dt;

        float y = 0.f;
        #pragma unroll
        for (int n = 0; n < D_STATE; ++n) {
            float dA = expf(dt * A[n]);
            h[n] = dA * h[n] + xd * s_proj[1 + n];
            y += h[n] * s_proj[1 + D_STATE + n];
        }
        yb[(size_t)l * D_INNER + d] = y + xv * Dp;
        __syncthreads();
    }
}

// ---------------------------------------------------------------------------
// LayerNorm over D_INNER + silu(z) gate; emit bf16 for the out_proj WMMA GEMM.
// One block per row m.
// ---------------------------------------------------------------------------
__global__ __launch_bounds__(256) void ln_gate_kernel(const float* __restrict__ y_raw,
                                                      const float* __restrict__ xz,
                                                      const float* __restrict__ ln_g,
                                                      const float* __restrict__ ln_b,
                                                      unsigned short* __restrict__ y_bf)
{
    __shared__ float s_sum[256];
    __shared__ float s_sq[256];
    const int m = blockIdx.x;
    const float* yr = y_raw + (size_t)m * D_INNER;
    const float* zr = xz + (size_t)m * N_IN + D_INNER;

    float s = 0.f, q = 0.f;
    for (int d = threadIdx.x; d < D_INNER; d += 256) {
        float v = yr[d];
        s += v; q += v * v;
    }
    s_sum[threadIdx.x] = s;
    s_sq[threadIdx.x]  = q;
    __syncthreads();
    for (int st = 128; st > 0; st >>= 1) {
        if (threadIdx.x < st) {
            s_sum[threadIdx.x] += s_sum[threadIdx.x + st];
            s_sq[threadIdx.x]  += s_sq[threadIdx.x + st];
        }
        __syncthreads();
    }
    const float mu  = s_sum[0] * (1.0f / D_INNER);
    const float var = s_sq[0] * (1.0f / D_INNER) - mu * mu;
    const float rstd = rsqrtf(var + LN_EPS);

    for (int d = threadIdx.x; d < D_INNER; d += 256) {
        float v = (yr[d] - mu) * rstd * ln_g[d] + ln_b[d];
        float z = zr[d];
        v *= siluf_(z);
        y_bf[(size_t)m * D_INNER + d] = f2bf(v);
    }
}

// ---------------------------------------------------------------------------
// Orchestration
// ---------------------------------------------------------------------------
extern "C" void kernel_launch(void* const* d_in, const int* in_sizes, int n_in,
                              void* d_out, int out_size, void* d_ws, size_t ws_size,
                              hipStream_t stream) {
    const float* x         = (const float*)d_in[0];   // [B,L,D_MODEL]
    const float* in_proj_w = (const float*)d_in[1];   // [4096,1024]
    const float* conv_w    = (const float*)d_in[2];   // [2048,4]
    const float* conv_b    = (const float*)d_in[3];   // [2048]
    const float* x_proj_w  = (const float*)d_in[4];   // [33,2048]
    const float* A_log     = (const float*)d_in[5];   // [2048,16]
    const float* D_param   = (const float*)d_in[6];   // [2048]
    const float* dt_w      = (const float*)d_in[7];   // [2048,1]
    const float* dt_b      = (const float*)d_in[8];   // [2048]
    const float* out_w     = (const float*)d_in[9];   // [1024,2048]
    const float* ln_g      = (const float*)d_in[10];  // [2048]
    const float* ln_b      = (const float*)d_in[11];  // [2048]
    float* out = (float*)d_out;                       // [B,L,D_MODEL] f32

    // Workspace carve-up (256-B aligned slices)
    char* ws = (char*)d_ws;
    size_t off = 0;
    auto carve = [&](size_t bytes) -> void* {
        void* p = ws + off;
        off = (off + bytes + 255) & ~(size_t)255;
        return p;
    };
    unsigned short* x_bf     = (unsigned short*)carve((size_t)MROWS * D_MODEL * 2);
    unsigned short* w_in_bf  = (unsigned short*)carve((size_t)N_IN * D_MODEL * 2);
    unsigned short* w_out_bf = (unsigned short*)carve((size_t)D_MODEL * D_INNER * 2);
    float*          xz       = (float*)carve((size_t)MROWS * N_IN * 4);
    float*          xb       = (float*)carve((size_t)MROWS * D_INNER * 4);
    float*          proj     = (float*)carve((size_t)MROWS * NPROJ * 4);
    float*          y_raw    = (float*)carve((size_t)MROWS * D_INNER * 4);
    unsigned short* y_bf     = (unsigned short*)carve((size_t)MROWS * D_INNER * 2);
    (void)ws_size; (void)in_sizes; (void)n_in; (void)out_size;

    // 1) bf16 conversions for WMMA operands
    {
        int n1 = MROWS * D_MODEL;
        cvt_bf16_kernel<<<(n1 + 255) / 256, 256, 0, stream>>>(x, x_bf, n1);
        int n2 = N_IN * D_MODEL;
        cvt_bf16_kernel<<<(n2 + 255) / 256, 256, 0, stream>>>(in_proj_w, w_in_bf, n2);
        int n3 = D_MODEL * D_INNER;
        cvt_bf16_kernel<<<(n3 + 255) / 256, 256, 0, stream>>>(out_w, w_out_bf, n3);
    }

    // 2) in_proj GEMM: xz[M, 4096] = x[M,1024] @ in_proj_w[4096,1024]^T
    gemm_bf16_wmma_kernel<<<dim3(N_IN / 64, MROWS / 64), 128, 0, stream>>>(
        x_bf, w_in_bf, xz, MROWS, N_IN, D_MODEL);

    // 3) causal conv + SiLU
    conv_silu_kernel<<<(MROWS * D_INNER) / 256, 256, 0, stream>>>(xz, conv_w, conv_b, xb);

    // 4) x_proj (skinny N=33): LDS-staged split-K
    xproj_kernel<<<MROWS, 256, 0, stream>>>(xb, x_proj_w, proj);

    // 5) selective scan
    scan_kernel<<<dim3(D_INNER / 256, BB), 256, 0, stream>>>(
        xb, proj, A_log, D_param, dt_w, dt_b, y_raw);

    // 6) LayerNorm + silu(z) gate -> bf16
    ln_gate_kernel<<<MROWS, 256, 0, stream>>>(y_raw, xz, ln_g, ln_b, y_bf);

    // 7) out_proj GEMM: out[M, 1024] = y[M,2048] @ out_w[1024,2048]^T
    gemm_bf16_wmma_kernel<<<dim3(D_MODEL / 64, MROWS / 64), 128, 0, stream>>>(
        y_bf, w_out_bf, out, MROWS, D_MODEL, D_INNER);
}
